// MultipleMILTransformer_44959717654918
// MI455X (gfx1250) — matrix operand order repaired and checked by reading, not compiled, not run on hardware
//
#include <hip/hip_runtime.h>

// ---------------- problem constants ----------------
constexpr int Dm    = 512;                 // model dim
constexpr int Hh    = 8;                   // heads
constexpr int Dh    = 64;                  // head dim
constexpr int Gg    = 16;                  // groups
constexpr int SEQ   = 1025;                // msg token + 1024 tokens per group
constexpr int SP    = 1040;                // SEQ padded to multiple of 16
constexpr int NROW  = Gg * SP;             // 16640 rows (multiple of 128)
constexpr int E3    = 3 * Dm;              // 1536
constexpr int NTOK  = 16384;
constexpr int INDIM = 1024;

typedef unsigned short u16;

// ---------------- wmma vector types ----------------
typedef __attribute__((ext_vector_type(16))) __bf16        v16bf;
typedef __attribute__((ext_vector_type(8)))  float         v8f;
typedef __attribute__((ext_vector_type(8)))  unsigned int  v8u;
typedef __attribute__((ext_vector_type(4)))  int           v4i;

__device__ __forceinline__ u16 f2bf(float f) {
    unsigned int u = __float_as_uint(f);
    unsigned int r = u + 0x7FFFu + ((u >> 16) & 1u);   // RNE to bf16
    return (u16)(r >> 16);
}

// ---- CDNA5 async global->LDS copy (ASYNCcnt) with portable fallback ----
#if __has_builtin(__builtin_amdgcn_global_load_async_to_lds_b128)
#define HAVE_ASYNC_LDS 1
#else
#define HAVE_ASYNC_LDS 0
#endif

__device__ __forceinline__ void cp_b128_to_lds(void* lds_dst, const void* gsrc) {
#if HAVE_ASYNC_LDS
    __builtin_amdgcn_global_load_async_to_lds_b128(
        (__attribute__((address_space(1))) v4i*)(void*)gsrc,
        (__attribute__((address_space(3))) v4i*)lds_dst, 0, 0);
#else
    *(uint4*)lds_dst = *(const uint4*)gsrc;
#endif
}
__device__ __forceinline__ void wait_lds_copies() {
#if HAVE_ASYNC_LDS
#if __has_builtin(__builtin_amdgcn_s_wait_asynccnt)
    __builtin_amdgcn_s_wait_asynccnt(0);
#else
    asm volatile("s_wait_asynccnt 0" ::: "memory");
#endif
#endif
}

// =====================================================================
// elementwise fp32 -> bf16
// =====================================================================
__global__ void f32_to_bf16_kernel(const float* __restrict__ in,
                                   u16* __restrict__ out, int n)
{
    int i = blockIdx.x * blockDim.x + threadIdx.x;
    if (i < n) out[i] = f2bf(in[i]);
}

// =====================================================================
// init: msg token rows, zero padding rows (fp32 master + bf16 shadow), cls
// =====================================================================
__global__ void init_groups_kernel(const float* __restrict__ msg_tok,
                                   const float* __restrict__ msgcls,
                                   float* groups_f, u16* groups_bf, float* cls)
{
    int idx = blockIdx.x * blockDim.x + threadIdx.x;
    const int total = Gg * 16 * Dm;            // per group: row 0 + 15 pad rows
    if (idx < total) {
        int d  = idx & (Dm - 1);
        int rr = (idx >> 9) & 15;              // 0..15
        int g  = idx >> 13;                    // / (512*16)
        int p  = (rr == 0) ? 0 : (1024 + rr);  // 0 or 1025..1039
        float v = (rr == 0) ? msg_tok[d] : 0.0f;
        size_t off = ((size_t)g * SP + p) * Dm + d;
        groups_f[off]  = v;
        groups_bf[off] = f2bf(v);
    }
    if (idx < Dm) cls[idx] = msgcls[idx];
}

// =====================================================================
// bf16 WMMA GEMM:  C[m,n] = sum_k A[m,k]*B[n,k] (+bias1[n]+bias2[n]) (+residual)
// A, B already bf16; LDS staging is a raw b128 copy (async-to-LDS on CDNA5).
// block = 256 threads (8 waves), tile 128x128, K-step 32.
// Outputs: fp32 (Cf) and/or bf16 (Cbf) -- both nullable.
// row_map_fc1: output row = g*SP + 1 + (m % 1024)   (fc1 -> grouped layout)
// mask_pad   : rows with (m % SP) >= SEQ are written as 0
// =====================================================================
__global__ __launch_bounds__(256)
void gemm_bf16_kernel(const u16* __restrict__ A, int lda,
                      const u16* __restrict__ B,
                      const float* __restrict__ bias1, const float* __restrict__ bias2,
                      const float* residual, float* Cf, u16* Cbf,
                      int M, int N, int K,
                      int row_map_fc1, int mask_pad)
{
    __shared__ __align__(16) u16 As[128 * 32];
    __shared__ __align__(16) u16 Bs[128 * 32];

    const int tid  = threadIdx.x;
    const int wave = tid >> 5;
    const int lane = tid & 31;
    const int hl   = lane >> 4;
    const int ln   = lane & 15;

    const int ntiles = N >> 7;
    const int m0 = (blockIdx.x / ntiles) << 7;
    const int n0 = (blockIdx.x % ntiles) << 7;

    const v8f zero = {0.f,0.f,0.f,0.f,0.f,0.f,0.f,0.f};
    v8f acc[8] = {zero, zero, zero, zero, zero, zero, zero, zero};

    const unsigned int* As32 = (const unsigned int*)As;
    const unsigned int* Bs32 = (const unsigned int*)Bs;

    for (int k0 = 0; k0 < K; k0 += 32) {
        // stage A (128x32 bf16) and B (128x32 bf16): pure b128 copies
        #pragma unroll
        for (int it = 0; it < 2; ++it) {
            int i = tid + it * 256;            // 0..511 b128 slots per matrix
            int r = i >> 2, c = i & 3;         // row, 8-element column
            cp_b128_to_lds(&As[r * 32 + c * 8], A + (size_t)(m0 + r) * lda + k0 + c * 8);
            cp_b128_to_lds(&Bs[r * 32 + c * 8], B + (size_t)(n0 + r) * K   + k0 + c * 8);
        }
        // prefetch next K tile (one line per row, 256 threads cover A+B rows)
        if (k0 + 32 < K) {
            int r = tid & 127;
            const u16* p = (tid < 128) ? (A + (size_t)(m0 + r) * lda + k0 + 32)
                                       : (B + (size_t)(n0 + r) * K   + k0 + 32);
            __builtin_prefetch(p, 0, 0);
        }
        wait_lds_copies();
        __syncthreads();

        // A fragment: lane row m = wave*16 + ln ; k = i + 16*hl
        v8u au;
        const int am = wave * 16 + ln;
        #pragma unroll
        for (int p = 0; p < 8; ++p) {
            int k = p * 2 + hl * 16;   // even pair base
            au[p] = As32[am * 16 + (k >> 1)];
        }
        v16bf af = __builtin_bit_cast(v16bf, au);

        // B fragments: lane col n = t*16 + ln ; k = i + 16*hl -> pair p + 8*hl
        #pragma unroll
        for (int t = 0; t < 8; ++t) {
            v8u bu;
            const int bn = t * 16 + ln;
            #pragma unroll
            for (int p = 0; p < 8; ++p)
                bu[p] = Bs32[bn * 16 + p + 8 * hl];
            v16bf bf = __builtin_bit_cast(v16bf, bu);
            acc[t] = __builtin_amdgcn_wmma_f32_16x16x32_bf16(
                        false, af, false, bf, (short)0, acc[t], false, false);
        }
        __syncthreads();
    }

    // ---------------- epilogue (bias hoisted per n-subtile) ----------------
    const bool has_res = residual != nullptr;
    #pragma unroll
    for (int t = 0; t < 8; ++t) {
        const int n = n0 + t * 16 + ln;
        float bv = 0.0f;
        if (bias1) bv += bias1[n];
        if (bias2) bv += bias2[n];
        #pragma unroll
        for (int r = 0; r < 8; ++r) {
            int m = m0 + wave * 16 + r + hl * 8;
            size_t orow;
            bool valid = true;
            if (row_map_fc1) {
                int g = m >> 10, p = m & 1023;
                orow = (size_t)g * SP + 1 + p;
            } else {
                orow = (size_t)m;
                if (mask_pad && (m % SP) >= SEQ) valid = false;
            }
            float outv = 0.0f;
            if (valid) {
                outv = acc[t][r] + bv;
                if (has_res) outv += residual[orow * (size_t)N + n];
            }
            size_t oidx = orow * (size_t)N + n;
            if (Cf)  Cf[oidx]  = outv;
            if (Cbf) Cbf[oidx] = f2bf(outv);
        }
    }
}

// =====================================================================
// Flash attention over one (group, head, 16-query tile). One wave / block.
// qkv is bf16: [G][SP][1536] (q|k|v); keys >= SEQ masked. Output o is bf16.
// V chunk staged transposed through LDS so fragments are ds_load_b128.
// =====================================================================
__global__ __launch_bounds__(32)
void flash_attn_kernel(const u16* __restrict__ qkv, u16* __restrict__ o)
{
    __shared__ __align__(16) u16 Plds[16 * 32];
    __shared__ __align__(16) u16 VT[64 * 32];   // [dh][key]

    const int lane = threadIdx.x;
    const int hl = lane >> 4, ln = lane & 15;
    const int q0 = blockIdx.x * 16;
    const int h  = blockIdx.y;
    const int g  = blockIdx.z;
    const size_t base = (size_t)g * SP * E3;
    const float scale = 0.125f;                // 1/sqrt(64)

    // Q fragments (A layout): row m = ln; contiguous 8-element bf16 runs
    v16bf qf[2];
    {
        const u16* qp = qkv + base + (size_t)(q0 + ln) * E3 + h * Dh;
        #pragma unroll
        for (int s = 0; s < 2; ++s) {
            v8u u;
            #pragma unroll
            for (int gi = 0; gi < 2; ++gi) {
                uint4 t = *(const uint4*)(qp + s * 32 + gi * 8 + hl * 16);
                u[gi * 4 + 0] = t.x; u[gi * 4 + 1] = t.y;
                u[gi * 4 + 2] = t.z; u[gi * 4 + 3] = t.w;
            }
            qf[s] = __builtin_bit_cast(v16bf, u);
        }
    }

    float rmax[8], rsum[8];
    #pragma unroll
    for (int r = 0; r < 8; ++r) { rmax[r] = -1e30f; rsum[r] = 0.0f; }
    const v8f zero = {0.f,0.f,0.f,0.f,0.f,0.f,0.f,0.f};
    v8f ov[4] = {zero, zero, zero, zero};

    const unsigned int* P32  = (const unsigned int*)Plds;
    const unsigned int* VT32 = (const unsigned int*)VT;

    for (int k0 = 0; k0 < SEQ; k0 += 32) {
        // ---- stage V chunk (32 keys x 64 dh, bf16) transposed into LDS ----
        #pragma unroll
        for (int it = 0; it < 8; ++it) {
            int i   = lane + it * 32;          // 0..255 b128 slots
            int row = i >> 3;                  // key 0..31
            int c   = i & 7;                   // 8-element dh column
            int gr  = k0 + row;
            int kr  = (gr < SP) ? gr : 0;
            uint4 t = *(const uint4*)(qkv + base + (size_t)kr * E3 + 2 * Dm + h * Dh + c * 8);
            int d0 = c * 8;
            VT[(d0 + 0) * 32 + row] = (u16)(t.x & 0xFFFF);
            VT[(d0 + 1) * 32 + row] = (u16)(t.x >> 16);
            VT[(d0 + 2) * 32 + row] = (u16)(t.y & 0xFFFF);
            VT[(d0 + 3) * 32 + row] = (u16)(t.y >> 16);
            VT[(d0 + 4) * 32 + row] = (u16)(t.z & 0xFFFF);
            VT[(d0 + 5) * 32 + row] = (u16)(t.z >> 16);
            VT[(d0 + 6) * 32 + row] = (u16)(t.w & 0xFFFF);
            VT[(d0 + 7) * 32 + row] = (u16)(t.w >> 16);
        }

        // ---- scores for 32 keys as two 16-key WMMA tiles ----
        float sv[2][8];
        #pragma unroll
        for (int t = 0; t < 2; ++t) {
            int kb = k0 + t * 16 + ln;                 // this lane's key column
            int kr = (kb < SP) ? kb : 0;
            const u16* kp = qkv + base + (size_t)kr * E3 + Dm + h * Dh;
            v8f sacc = zero;
            #pragma unroll
            for (int s = 0; s < 2; ++s) {
                v8u u;                                  // B layout: d = s*32 + i + 16*hl
                uint4 a = *(const uint4*)(kp + s * 32 + 16 * hl);
                uint4 b = *(const uint4*)(kp + s * 32 + 16 * hl + 8);
                u[0] = a.x; u[1] = a.y; u[2] = a.z; u[3] = a.w;
                u[4] = b.x; u[5] = b.y; u[6] = b.z; u[7] = b.w;
                v16bf kf = __builtin_bit_cast(v16bf, u);
                sacc = __builtin_amdgcn_wmma_f32_16x16x32_bf16(
                          false, qf[s], false, kf, (short)0, sacc, false, false);
            }
            bool valid = kb < SEQ;
            #pragma unroll
            for (int r = 0; r < 8; ++r)
                sv[t][r] = valid ? sacc[r] * scale : -1e30f;
        }

        // ---- online softmax update (row stats per half-wave) ----
        float alpha[8], p0[8], p1[8];
        #pragma unroll
        for (int r = 0; r < 8; ++r) {
            float m = fmaxf(sv[0][r], sv[1][r]);
            #pragma unroll
            for (int off = 1; off < 16; off <<= 1)
                m = fmaxf(m, __shfl_xor(m, off, 32));
            float nm = fmaxf(rmax[r], m);
            alpha[r] = __expf(rmax[r] - nm);
            rmax[r]  = nm;
            p0[r] = __expf(sv[0][r] - nm);
            p1[r] = __expf(sv[1][r] - nm);
            float s = p0[r] + p1[r];
            #pragma unroll
            for (int off = 1; off < 16; off <<= 1)
                s += __shfl_xor(s, off, 32);
            rsum[r] = rsum[r] * alpha[r] + s;
        }
        #pragma unroll
        for (int t4 = 0; t4 < 4; ++t4)
            #pragma unroll
            for (int r = 0; r < 8; ++r) ov[t4][r] *= alpha[r];

        // ---- transpose P (C layout -> A layout) through LDS ----
        #pragma unroll
        for (int r = 0; r < 8; ++r) {
            int m = r + hl * 8;
            Plds[m * 32 + ln]      = f2bf(p0[r]);
            Plds[m * 32 + 16 + ln] = f2bf(p1[r]);
        }
        __syncthreads();
        v8u pu;
        #pragma unroll
        for (int p = 0; p < 8; ++p) {
            int k = p * 2 + hl * 16;
            pu[p] = P32[ln * 16 + (k >> 1)];
        }
        v16bf pf = __builtin_bit_cast(v16bf, pu);

        // ---- O += P @ V  (4 dh sub-tiles, V fragments from transposed LDS) ----
        #pragma unroll
        for (int t4 = 0; t4 < 4; ++t4) {
            v8u vu;                                     // B layout: key k = i + 16*hl
            #pragma unroll
            for (int p = 0; p < 8; ++p)
                vu[p] = VT32[(t4 * 16 + ln) * 16 + p + 8 * hl];
            v16bf vf = __builtin_bit_cast(v16bf, vu);
            ov[t4] = __builtin_amdgcn_wmma_f32_16x16x32_bf16(
                        false, pf, false, vf, (short)0, ov[t4], false, false);
        }
        __syncthreads();
    }

    // ---- normalize and store bf16 (zeros for padded query rows) ----
    #pragma unroll
    for (int t4 = 0; t4 < 4; ++t4) {
        #pragma unroll
        for (int r = 0; r < 8; ++r) {
            int m = r + hl * 8;
            int q = q0 + m;
            float val = (q < SEQ) ? ov[t4][r] / rsum[r] : 0.0f;
            o[((size_t)g * SP + q) * Dm + h * Dh + t4 * 16 + ln] = f2bf(val);
        }
    }
}

// =====================================================================
// message-attention helpers (tiny: seq 17, fp32)
// =====================================================================
__global__ void gather_mseq_kernel(const float* __restrict__ cls,
                                   const float* __restrict__ groups,
                                   float* __restrict__ mseq)
{
    int idx = blockIdx.x * blockDim.x + threadIdx.x;
    if (idx >= 17 * Dm) return;
    int row = idx >> 9, d = idx & (Dm - 1);
    mseq[idx] = (row == 0) ? cls[d] : groups[((size_t)(row - 1) * SP) * Dm + d];
}

__global__ void small_gemm_kernel(const float* __restrict__ A, const float* __restrict__ B,
                                  const float* __restrict__ bias, const float* __restrict__ residual,
                                  float* __restrict__ C, int M, int N, int K)
{
    int idx = blockIdx.x * blockDim.x + threadIdx.x;
    if (idx >= M * N) return;
    int m = idx / N, n = idx % N;
    float acc = 0.0f;
    for (int k = 0; k < K; ++k) acc += A[m * K + k] * B[n * K + k];
    if (bias)     acc += bias[n];
    if (residual) acc += residual[idx];
    C[idx] = acc;
}

__global__ __launch_bounds__(64)
void small_attn_kernel(const float* __restrict__ mqkv, float* __restrict__ mo)
{
    __shared__ float q[17][64], k[17][64], v[17][64], p[17][17];
    const int h = blockIdx.x, tid = threadIdx.x;
    for (int i = tid; i < 17 * 64; i += 64) {
        int r = i >> 6, d = i & 63;
        q[r][d] = mqkv[r * E3 + h * Dh + d];
        k[r][d] = mqkv[r * E3 + Dm + h * Dh + d];
        v[r][d] = mqkv[r * E3 + 2 * Dm + h * Dh + d];
    }
    __syncthreads();
    for (int i = tid; i < 17 * 17; i += 64) {
        int r = i / 17, c = i % 17;
        float s = 0.0f;
        for (int d = 0; d < 64; ++d) s += q[r][d] * k[c][d];
        p[r][c] = s * 0.125f;
    }
    __syncthreads();
    if (tid < 17) {
        float mx = -1e30f;
        for (int c = 0; c < 17; ++c) mx = fmaxf(mx, p[tid][c]);
        float sum = 0.0f;
        for (int c = 0; c < 17; ++c) { float e = __expf(p[tid][c] - mx); p[tid][c] = e; sum += e; }
        float inv = 1.0f / sum;
        for (int c = 0; c < 17; ++c) p[tid][c] *= inv;
    }
    __syncthreads();
    for (int i = tid; i < 17 * 64; i += 64) {
        int r = i >> 6, d = i & 63;
        float acc = 0.0f;
        for (int c = 0; c < 17; ++c) acc += p[r][c] * v[c][d];
        mo[r * Dm + h * Dh + d] = acc;
    }
}

__global__ void scatter_mseq_kernel(const float* __restrict__ mseq2,
                                    float* __restrict__ cls,
                                    float* __restrict__ groups_f,
                                    u16* __restrict__ groups_bf)
{
    int idx = blockIdx.x * blockDim.x + threadIdx.x;
    if (idx >= 17 * Dm) return;
    int row = idx >> 9, d = idx & (Dm - 1);
    float v = mseq2[idx];
    if (row == 0) cls[d] = v;
    else {
        size_t off = ((size_t)(row - 1) * SP) * Dm + d;
        groups_f[off]  = v;
        groups_bf[off] = f2bf(v);
    }
}

__global__ __launch_bounds__(32)
void fc2_softmax_kernel(const float* __restrict__ cls, const float* __restrict__ w,
                        const float* __restrict__ b, float* __restrict__ out)
{
    int lane = threadIdx.x;
    float a0 = 0.0f, a1 = 0.0f;
    for (int k = lane; k < Dm; k += 32) {
        float c = cls[k];
        a0 += c * w[k];
        a1 += c * w[Dm + k];
    }
    for (int off = 16; off; off >>= 1) {
        a0 += __shfl_xor(a0, off, 32);
        a1 += __shfl_xor(a1, off, 32);
    }
    if (lane == 0) {
        float l0 = a0 + b[0], l1 = a1 + b[1];
        float m = fmaxf(l0, l1);
        float e0 = __expf(l0 - m), e1 = __expf(l1 - m);
        float inv = 1.0f / (e0 + e1);
        out[0] = l0; out[1] = l1; out[2] = e0 * inv; out[3] = e1 * inv;
    }
}

// =====================================================================
// host launcher
// =====================================================================
extern "C" void kernel_launch(void* const* d_in, const int* in_sizes, int n_in,
                              void* d_out, int out_size, void* d_ws, size_t ws_size,
                              hipStream_t stream)
{
    (void)in_sizes; (void)n_in; (void)out_size; (void)ws_size;
    const float* x       = (const float*)d_in[0];
    const float* fc1_w   = (const float*)d_in[1];
    const float* fc1_b   = (const float*)d_in[2];
    const float* pos_emb = (const float*)d_in[3];
    const float* msg_tok = (const float*)d_in[4];
    const float* msgcls  = (const float*)d_in[5];
    const float* gqkv    = (const float*)d_in[6];
    const float* gout_w  = (const float*)d_in[7];
    const float* gout_b  = (const float*)d_in[8];
    const float* mqkv    = (const float*)d_in[9];
    const float* mout_w  = (const float*)d_in[10];
    const float* mout_b  = (const float*)d_in[11];
    const float* fc2_w   = (const float*)d_in[12];
    const float* fc2_b   = (const float*)d_in[13];
    float* out = (float*)d_out;

    // ---------------- workspace layout (~160 MB) ----------------
    float* groups_f = (float*)d_ws;                        // [G][SP][D] fp32 master
    float* cls      = groups_f + (size_t)NROW * Dm;
    float* mseq     = cls   + Dm;
    float* mqkvb    = mseq  + 17 * Dm;
    float* mo       = mqkvb + 17 * E3;
    float* mseq2    = mo    + 17 * Dm;
    u16* x_bf       = (u16*)(mseq2 + 17 * Dm);             // [NTOK][INDIM]
    u16* wfc1_bf    = x_bf     + (size_t)NTOK * INDIM;     // [D][INDIM]
    u16* wqkv_bf    = wfc1_bf  + (size_t)Dm * INDIM;       // [L][3D][D]
    u16* wout_bf    = wqkv_bf  + (size_t)2 * E3 * Dm;      // [L][D][D]
    u16* groups_bf  = wout_bf  + (size_t)2 * Dm * Dm;      // [G][SP][D]
    u16* qkv_bf     = groups_bf + (size_t)NROW * Dm;       // [G][SP][3D]
    u16* o_bf       = qkv_bf   + (size_t)NROW * E3;        // [G][SP][D]

    // one-time fp32 -> bf16 conversions (inputs + weights)
    auto cvt = [&](const float* in, u16* o_, int n) {
        f32_to_bf16_kernel<<<(n + 255) / 256, 256, 0, stream>>>(in, o_, n);
    };
    cvt(x,      x_bf,    NTOK * INDIM);
    cvt(fc1_w,  wfc1_bf, Dm * INDIM);
    cvt(gqkv,   wqkv_bf, 2 * E3 * Dm);
    cvt(gout_w, wout_bf, 2 * Dm * Dm);

    init_groups_kernel<<<512, 256, 0, stream>>>(msg_tok, msgcls, groups_f, groups_bf, cls);

    // fc1 + pos_embed, scattered into grouped layout (rows 1..1024 of each group)
    gemm_bf16_kernel<<<(NTOK / 128) * (Dm / 128), 256, 0, stream>>>(
        x_bf, INDIM, wfc1_bf, fc1_b, pos_emb, nullptr, groups_f, groups_bf,
        NTOK, Dm, INDIM, /*row_map_fc1=*/1, /*mask_pad=*/0);

    for (int l = 0; l < 2; ++l) {
        // qkv projection (bf16 out only; padding rows are zero -> zero qkv)
        gemm_bf16_kernel<<<(NROW / 128) * (E3 / 128), 256, 0, stream>>>(
            groups_bf, Dm, wqkv_bf + (size_t)l * E3 * Dm, nullptr, nullptr,
            nullptr, nullptr, qkv_bf, NROW, E3, Dm, 0, 0);

        // flash attention per (group, head, 16-query tile)
        flash_attn_kernel<<<dim3(SP / 16, Hh, Gg), 32, 0, stream>>>(qkv_bf, o_bf);

        // out projection + bias + residual(fp32 master); padding rows zeroed
        gemm_bf16_kernel<<<(NROW / 128) * (Dm / 128), 256, 0, stream>>>(
            o_bf, Dm, wout_bf + (size_t)l * Dm * Dm, gout_b + (size_t)l * Dm, nullptr,
            groups_f, groups_f, groups_bf, NROW, Dm, Dm, 0, /*mask_pad=*/1);

        // message attention over [cls; 16 msg tokens] (fp32, tiny)
        gather_mseq_kernel<<<(17 * Dm + 255) / 256, 256, 0, stream>>>(cls, groups_f, mseq);
        small_gemm_kernel<<<(17 * E3 + 255) / 256, 256, 0, stream>>>(
            mseq, mqkv + (size_t)l * E3 * Dm, nullptr, nullptr, mqkvb, 17, E3, Dm);
        small_attn_kernel<<<Hh, 64, 0, stream>>>(mqkvb, mo);
        small_gemm_kernel<<<(17 * Dm + 255) / 256, 256, 0, stream>>>(
            mo, mout_w + (size_t)l * Dm * Dm, mout_b + (size_t)l * Dm, mseq, mseq2,
            17, Dm, Dm);
        scatter_mseq_kernel<<<(17 * Dm + 255) / 256, 256, 0, stream>>>(
            mseq2, cls, groups_f, groups_bf);
    }

    fc2_softmax_kernel<<<1, 32, 0, stream>>>(cls, fc2_w, fc2_b, out);
}
